// PixtralHFAttention_80401787781161
// MI455X (gfx1250) — compile-verified, matrix operand order checked by reading, not run on hardware
//
#include <hip/hip_runtime.h>
#include <hip/hip_bf16.h>

// ---------------------------------------------------------------------------
// Pixtral attention for gfx1250 (MI455X): bf16 WMMA everywhere, fp32 accum,
// async global->LDS staging (ASYNCcnt) with double-buffered GEMM K-loop.
// B=2, S=2048, HIDDEN=1024, H=16, D=64.
// ---------------------------------------------------------------------------

#define BATCH   2
#define SEQ     2048
#define HID     1024
#define NH      16
#define HD      64
#define MTOT    (BATCH * SEQ)        // 4096 rows for the GEMMs

typedef __bf16 v16bf __attribute__((ext_vector_type(16)));
typedef float  v8f   __attribute__((ext_vector_type(8)));

union ABf16 { v16bf v; uint4 u[2]; };

static __device__ inline unsigned short f32_to_bf16(float f) {
    unsigned int u = __float_as_uint(f);
    u = u + 0x7FFFu + ((u >> 16) & 1u);        // round-to-nearest-even
    return (unsigned short)(u >> 16);
}
static __device__ inline float bf16_to_f32(unsigned short b) {
    return __uint_as_float(((unsigned int)b) << 16);
}

static __device__ inline v8f wmma_bf16(const ABf16& a, const ABf16& b, v8f c) {
    return __builtin_amdgcn_wmma_f32_16x16x32_bf16(
        /*neg_a=*/false, a.v, /*neg_b=*/false, b.v,
        /*c_mod=*/(short)0, c, /*reuse_a=*/false, /*reuse_b=*/false);
}

// ---- CDNA5 async global->LDS (16B per lane), tracked by ASYNCcnt ----------
// LDS byte address = low 32 bits of the flat pointer (ISA 10.2: LDS aperture
// selects on addr[63:32]).
static __device__ inline void async_load_b128(void* lds_ptr, const void* gptr) {
    unsigned lds_off = (unsigned)(unsigned long long)lds_ptr;
    unsigned long long ga = (unsigned long long)gptr;
    asm volatile("global_load_async_to_lds_b128 %0, %1, off"
                 :: "v"(lds_off), "v"(ga)
                 : "memory");
}
static __device__ inline void wait_async0(void) {
    asm volatile("s_wait_asynccnt 0x0" ::: "memory");
}

// ---------------------------------------------------------------------------
// fp32 -> bf16 bulk convert
// ---------------------------------------------------------------------------
__global__ __launch_bounds__(256) void cvt_bf16_kernel(const float* __restrict__ in,
                                                       unsigned short* __restrict__ out,
                                                       int n) {
    int i = blockIdx.x * 256 + threadIdx.x;
    if (i < n) out[i] = f32_to_bf16(in[i]);
}

// ---------------------------------------------------------------------------
// Tiled GEMM: C[M,N] = A[M,K] * B[N,K]^T   (both operands K-contiguous, bf16)
// 128x128 tile / WG, 8 waves (2Mx4N), each wave: 4x2 grid of 16x16 wmma tiles.
// Async double-buffered staging: DMA tile k+1 while WMMAs consume tile k.
// ---------------------------------------------------------------------------
template <bool OUT_F32>
__global__ __launch_bounds__(256)
void gemm_bf16_kernel(const unsigned short* __restrict__ A,
                      const unsigned short* __restrict__ Bw,
                      void* __restrict__ Cout,
                      int M, int N, int K) {
    __shared__ __align__(16) unsigned short As[2][128][32];
    __shared__ __align__(16) unsigned short Bs[2][128][32];

    const int t    = threadIdx.x;
    const int lane = t & 31;
    const int wv   = t >> 5;
    const int wm   = wv & 1;          // 0..1  (64 rows each)
    const int wn   = wv >> 1;         // 0..3  (32 cols each)
    const int lr   = lane & 15;
    const int kb   = (lane >> 4) * 8; // K sub-chunk base per lane-half
    const int rh   = (lane >> 4) * 8; // M offset for C rows per lane-half

    const int mBlk = blockIdx.x * 128;
    const int nBlk = blockIdx.y * 128;

    // staging assignment: 512 x 16B chunks per tile, 2 per thread
    const int srow0 = t >> 2;                    // chunk t
    const int soff0 = (t & 3) * 8;
    const int srow1 = (t + 256) >> 2;            // chunk t+256
    const int soff1 = ((t + 256) & 3) * 8;

    v8f acc[4][2] = {};

    // prologue: DMA tile 0
    async_load_b128(&As[0][srow0][soff0], &A[(size_t)(mBlk + srow0) * K + soff0]);
    async_load_b128(&Bs[0][srow0][soff0], &Bw[(size_t)(nBlk + srow0) * K + soff0]);
    async_load_b128(&As[0][srow1][soff1], &A[(size_t)(mBlk + srow1) * K + soff1]);
    async_load_b128(&Bs[0][srow1][soff1], &Bw[(size_t)(nBlk + srow1) * K + soff1]);
    wait_async0();
    __syncthreads();

    int buf = 0;
    for (int k0 = 0; k0 < K; k0 += 32) {
        // ---- issue async DMA for next K tile into the other buffer ----
        if (k0 + 32 < K) {
            const int nb = buf ^ 1, kn = k0 + 32;
            async_load_b128(&As[nb][srow0][soff0], &A[(size_t)(mBlk + srow0) * K + kn + soff0]);
            async_load_b128(&Bs[nb][srow0][soff0], &Bw[(size_t)(nBlk + srow0) * K + kn + soff0]);
            async_load_b128(&As[nb][srow1][soff1], &A[(size_t)(mBlk + srow1) * K + kn + soff1]);
            async_load_b128(&Bs[nb][srow1][soff1], &Bw[(size_t)(nBlk + srow1) * K + kn + soff1]);
        }

        // ---- fragments from LDS (ISA A/B layout: row per lane, K split) ----
        ABf16 afr[4], bfr[2];
        #pragma unroll
        for (int mi = 0; mi < 4; ++mi) {
            int r = wm * 64 + mi * 16 + lr;
            afr[mi].u[0] = *(const uint4*)&As[buf][r][kb];
            afr[mi].u[1] = *(const uint4*)&As[buf][r][kb + 16];
        }
        #pragma unroll
        for (int ni = 0; ni < 2; ++ni) {
            int r = wn * 32 + ni * 16 + lr;
            bfr[ni].u[0] = *(const uint4*)&Bs[buf][r][kb];
            bfr[ni].u[1] = *(const uint4*)&Bs[buf][r][kb + 16];
        }
        #pragma unroll
        for (int mi = 0; mi < 4; ++mi)
            #pragma unroll
            for (int ni = 0; ni < 2; ++ni)
                acc[mi][ni] = wmma_bf16(afr[mi], bfr[ni], acc[mi][ni]);

        // next tile's DMA must land before anyone reads it; barrier flips buf
        wait_async0();
        __syncthreads();
        buf ^= 1;
    }

    // ---- store: C layout lane=N, vgpr r = M (M+8 for upper lane half) ----
    #pragma unroll
    for (int mi = 0; mi < 4; ++mi) {
        #pragma unroll
        for (int ni = 0; ni < 2; ++ni) {
            int n = nBlk + wn * 32 + ni * 16 + lr;
            #pragma unroll
            for (int r = 0; r < 8; ++r) {
                int m = mBlk + wm * 64 + mi * 16 + rh + r;
                float v = acc[mi][ni][r];
                if (OUT_F32)
                    ((float*)Cout)[(size_t)m * N + n] = v;
                else
                    ((unsigned short*)Cout)[(size_t)m * N + n] = f32_to_bf16(v);
            }
        }
    }
}

// ---------------------------------------------------------------------------
// RoPE + reshape:  Qg/Kg/Vg [M, HID] -> Qr/Kr [B,H,S,D] bf16, Vt [B,H,D,S] bf16
// rotate_half: out[0:32] = x1*cos - x2*sin ; out[32:64] = x2*cos + x1*sin
// ---------------------------------------------------------------------------
__global__ __launch_bounds__(256)
void rope_reshape_kernel(const unsigned short* __restrict__ Qg,
                         const unsigned short* __restrict__ Kg,
                         const unsigned short* __restrict__ Vg,
                         const float* __restrict__ cosT,
                         const float* __restrict__ sinT,
                         unsigned short* __restrict__ Qr,
                         unsigned short* __restrict__ Kr,
                         unsigned short* __restrict__ Vt) {
    int idx = blockIdx.x * 256 + threadIdx.x;      // over B*S*H*D = 4M
    int d = idx & (HD - 1);
    int h = (idx >> 6) & (NH - 1);
    int s = (idx >> 10) & (SEQ - 1);
    int b = idx >> 21;

    size_t mn  = (size_t)(b * SEQ + s) * HID + h * HD + d;
    int    dp  = (d < 32) ? d + 32 : d - 32;
    size_t mnp = (size_t)(b * SEQ + s) * HID + h * HD + dp;

    float c  = cosT[s * HD + d];
    float sn = sinT[s * HD + d];

    float q  = bf16_to_f32(Qg[mn]),  qp = bf16_to_f32(Qg[mnp]);
    float k  = bf16_to_f32(Kg[mn]),  kp = bf16_to_f32(Kg[mnp]);
    float rq = (d < 32) ? (q * c - qp * sn) : (q * c + qp * sn);
    float rk = (d < 32) ? (k * c - kp * sn) : (k * c + kp * sn);

    size_t bhsd = (((size_t)(b * NH + h) * SEQ) + s) * HD + d;
    Qr[bhsd] = f32_to_bf16(rq);
    Kr[bhsd] = f32_to_bf16(rk);
    size_t bhds = (((size_t)(b * NH + h) * HD) + d) * SEQ + s;
    Vt[bhds] = f32_to_bf16(bf16_to_f32(Vg[mn]));
}

// ---------------------------------------------------------------------------
// Flash attention: one WG = 64 queries of one (b,h); loop 32 KV blocks of 64.
// 8 waves: wave w -> query tile (w&3), kv/d tile pair (w>>2)*2+{0,1}.
// K/Vt tiles staged with async DMA each block.
// ---------------------------------------------------------------------------
__global__ __launch_bounds__(256)
void attn_kernel(const unsigned short* __restrict__ Qr,
                 const unsigned short* __restrict__ Kr,
                 const unsigned short* __restrict__ Vt,
                 const float* __restrict__ mask,
                 unsigned short* __restrict__ AO) {
    __shared__ __align__(16) unsigned short Qs[64][64];
    __shared__ __align__(16) unsigned short Ks[64][64];
    __shared__ __align__(16) unsigned short Vts[64][64];
    __shared__ __align__(16) unsigned short Pb[64][64];
    __shared__ __align__(16) float          Sf[64][64];
    __shared__ float rowm[64], rowl[64], alpha[64];

    const int t    = threadIdx.x;
    const int lane = t & 31;
    const int wv   = t >> 5;
    const int lr   = lane & 15;
    const int kb   = (lane >> 4) * 8;
    const int rh   = (lane >> 4) * 8;
    const int qi   = wv & 3;          // query sub-tile 0..3
    const int pp   = wv >> 2;         // pair index 0..1 (kv tiles / d tiles)

    const int qb = blockIdx.x * 64;   // query block base
    const int bh = blockIdx.y;        // b*NH + h
    const int b  = bh >> 4, h = bh & (NH - 1);
    const size_t qkbase = (size_t)bh * SEQ * HD;

    // staging assignment: 512 x 16B chunks per 64x64 bf16 tile, 2 per thread
    const int srow0 = t >> 3,         soff0 = (t & 7) * 8;
    const int srow1 = (t + 256) >> 3, soff1 = ((t + 256) & 7) * 8;

    // ---- async DMA Q tile (8KB) ----
    async_load_b128(&Qs[srow0][soff0], &Qr[qkbase + (size_t)(qb + srow0) * HD + soff0]);
    async_load_b128(&Qs[srow1][soff1], &Qr[qkbase + (size_t)(qb + srow1) * HD + soff1]);
    if (t < 64) { rowm[t] = -1e30f; rowl[t] = 0.0f; }
    v8f oacc[2] = {};
    wait_async0();
    __syncthreads();

    for (int j = 0; j < SEQ / 64; ++j) {
        const int kvb = j * 64;
        // ---- async DMA K tile [kv,d] and Vt tile [d,kv] ----
        async_load_b128(&Ks[srow0][soff0],  &Kr[qkbase + (size_t)(kvb + srow0) * HD + soff0]);
        async_load_b128(&Ks[srow1][soff1],  &Kr[qkbase + (size_t)(kvb + srow1) * HD + soff1]);
        async_load_b128(&Vts[srow0][soff0], &Vt[qkbase + (size_t)srow0 * SEQ + kvb + soff0]);
        async_load_b128(&Vts[srow1][soff1], &Vt[qkbase + (size_t)srow1 * SEQ + kvb + soff1]);
        wait_async0();
        __syncthreads();

        // ---- S = (Q K^T) * scale ----
        #pragma unroll
        for (int kt = 0; kt < 2; ++kt) {
            const int kvt = pp * 2 + kt;
            v8f s = {};
            #pragma unroll
            for (int d0 = 0; d0 < HD; d0 += 32) {
                ABf16 af, bf;
                af.u[0] = *(const uint4*)&Qs[qi * 16 + lr][d0 + kb];
                af.u[1] = *(const uint4*)&Qs[qi * 16 + lr][d0 + kb + 16];
                bf.u[0] = *(const uint4*)&Ks[kvt * 16 + lr][d0 + kb];
                bf.u[1] = *(const uint4*)&Ks[kvt * 16 + lr][d0 + kb + 16];
                s = wmma_bf16(af, bf, s);
            }
            #pragma unroll
            for (int r = 0; r < 8; ++r)
                Sf[qi * 16 + rh + r][kvt * 16 + lr] = s[r] * 0.125f;  // 1/sqrt(64)
        }
        __syncthreads();

        // ---- online softmax over this 64-wide block (+ attention mask) ----
        if (t < 64) {
            const int r = t;
            const float* mrow = &mask[((size_t)b * SEQ + (qb + r)) * SEQ + kvb];
            float mx = rowm[r];
            for (int c2 = 0; c2 < 64; ++c2) {
                float v = Sf[r][c2] + mrow[c2];
                Sf[r][c2] = v;
                mx = fmaxf(mx, v);
            }
            float al = __expf(rowm[r] - mx);
            float sum = 0.0f;
            for (int c2 = 0; c2 < 64; ++c2) {
                float p = __expf(Sf[r][c2] - mx);
                Pb[r][c2] = f32_to_bf16(p);
                sum += p;
            }
            rowl[r] = rowl[r] * al + sum;
            rowm[r] = mx;
            alpha[r] = al;
        }
        __syncthreads();

        // ---- rescale running O, then O += P @ V ----
        #pragma unroll
        for (int r = 0; r < 8; ++r) {
            float al = alpha[qi * 16 + rh + r];
            oacc[0][r] *= al;
            oacc[1][r] *= al;
        }
        #pragma unroll
        for (int dt = 0; dt < 2; ++dt) {
            const int dtile = pp * 2 + dt;
            #pragma unroll
            for (int kv0 = 0; kv0 < 64; kv0 += 32) {
                ABf16 af, bf;
                af.u[0] = *(const uint4*)&Pb[qi * 16 + lr][kv0 + kb];
                af.u[1] = *(const uint4*)&Pb[qi * 16 + lr][kv0 + kb + 16];
                bf.u[0] = *(const uint4*)&Vts[dtile * 16 + lr][kv0 + kb];
                bf.u[1] = *(const uint4*)&Vts[dtile * 16 + lr][kv0 + kb + 16];
                oacc[dt] = wmma_bf16(af, bf, oacc[dt]);
            }
        }
        __syncthreads();
    }

    // ---- normalize and write [B, S, H*D] bf16 ----
    #pragma unroll
    for (int dt = 0; dt < 2; ++dt) {
        int d = pp * 32 + dt * 16 + lr;
        #pragma unroll
        for (int r = 0; r < 8; ++r) {
            int q = qi * 16 + rh + r;
            float v = oacc[dt][r] / rowl[q];
            AO[((size_t)(b * SEQ) + qb + q) * HID + h * HD + d] = f32_to_bf16(v);
        }
    }
}

// ---------------------------------------------------------------------------
// Host-side orchestration
// ---------------------------------------------------------------------------
extern "C" void kernel_launch(void* const* d_in, const int* in_sizes, int n_in,
                              void* d_out, int out_size, void* d_ws, size_t ws_size,
                              hipStream_t stream) {
    const float* x    = (const float*)d_in[0];   // [B,S,HID]
    const float* mask = (const float*)d_in[1];   // [B,1,S,S]
    const float* cosT = (const float*)d_in[2];   // [S,HD]
    const float* sinT = (const float*)d_in[3];   // [S,HD]
    const float* wq   = (const float*)d_in[4];   // [HID,HID]
    const float* wk   = (const float*)d_in[5];
    const float* wv   = (const float*)d_in[6];
    const float* wo   = (const float*)d_in[7];

    char* ws = (char*)d_ws;
    size_t off = 0;
    auto take = [&](size_t bytes) -> char* {
        char* p = ws + off;
        off += (bytes + 255) & ~(size_t)255;
        return p;
    };
    const size_t XB = (size_t)MTOT * HID * 2;    // bf16 [4096,1024] = 8MB
    const size_t WB = (size_t)HID * HID * 2;     // bf16 [1024,1024] = 2MB

    unsigned short* Xb  = (unsigned short*)take(XB);
    unsigned short* Wqb = (unsigned short*)take(WB);
    unsigned short* Wkb = (unsigned short*)take(WB);
    unsigned short* Wvb = (unsigned short*)take(WB);
    unsigned short* Wob = (unsigned short*)take(WB);
    unsigned short* Qg  = (unsigned short*)take(XB);
    unsigned short* Kg  = (unsigned short*)take(XB);
    unsigned short* Vg  = (unsigned short*)take(XB);
    unsigned short* Qr  = (unsigned short*)take(XB);
    unsigned short* Kr  = (unsigned short*)take(XB);
    unsigned short* Vtb = (unsigned short*)take(XB);
    unsigned short* AO  = (unsigned short*)take(XB);
    (void)ws_size; (void)in_sizes; (void)n_in; (void)out_size;

    const int nX = MTOT * HID;       // 4,194,304
    const int nW = HID * HID;        // 1,048,576
    cvt_bf16_kernel<<<nX / 256, 256, 0, stream>>>(x,  Xb,  nX);
    cvt_bf16_kernel<<<nW / 256, 256, 0, stream>>>(wq, Wqb, nW);
    cvt_bf16_kernel<<<nW / 256, 256, 0, stream>>>(wk, Wkb, nW);
    cvt_bf16_kernel<<<nW / 256, 256, 0, stream>>>(wv, Wvb, nW);
    cvt_bf16_kernel<<<nW / 256, 256, 0, stream>>>(wo, Wob, nW);

    dim3 ggrid(MTOT / 128, HID / 128);           // 32 x 8
    gemm_bf16_kernel<false><<<ggrid, 256, 0, stream>>>(Xb, Wqb, Qg, MTOT, HID, HID);
    gemm_bf16_kernel<false><<<ggrid, 256, 0, stream>>>(Xb, Wkb, Kg, MTOT, HID, HID);
    gemm_bf16_kernel<false><<<ggrid, 256, 0, stream>>>(Xb, Wvb, Vg, MTOT, HID, HID);

    rope_reshape_kernel<<<nX / 256, 256, 0, stream>>>(Qg, Kg, Vg, cosT, sinT,
                                                      Qr, Kr, Vtb);

    dim3 agrid(SEQ / 64, BATCH * NH);            // 32 x 32
    attn_kernel<<<agrid, 256, 0, stream>>>(Qr, Kr, Vtb, mask, AO);

    gemm_bf16_kernel<true><<<ggrid, 256, 0, stream>>>(AO, Wob, d_out, MTOT, HID, HID);
}
